// FunctionalMapNet_26173530702518
// MI455X (gfx1250) — matrix operand order repaired and checked by reading, not compiled, not run on hardware
//
#include <hip/hip_runtime.h>
#include <hip/hip_bf16.h>

typedef __attribute__((ext_vector_type(16))) __bf16 v16bf;
typedef __attribute__((ext_vector_type(8)))  float  v8f;
typedef __attribute__((ext_vector_type(2)))  float  v2f;

constexpr int B_     = 16;
constexpr int V_     = 5000;
constexpr int NF_    = 352;   // 22 * 16 = 11 * 32
constexpr int K_     = 120;
constexpr int KP_    = 128;   // K padded to 8 * 16
constexpr int NSPLIT = 4;     // split-K chunks over V (5000 / 4 = 1250 each)

__device__ __forceinline__ void bf16_split(float x, __bf16& hi, __bf16& lo) {
  hi = (__bf16)x;
  lo = (__bf16)(x - (float)hi);
}

// ---------------------------------------------------------------------------
// Kernel 1: partial F_hat = feat^T (NFxV) * evecs (VxK) over one V-chunk.
// One wave computes a 32(M) x 128(N) strip (2 M-tiles share the B fragments).
// Split-bf16: a*b ~= a_hi*b_hi + a_hi*b_lo + a_lo*b_hi  -> 6 wmma / (tilepair,step)
// A fragment layout (16-bit A 16x32, ISA 7.12.2):
//   m = lane&15, g = lane>>4; VGPR j holds K = 2j + (j>=4?8:0) + 8g, and +1
// B fragment (32x16): n = lane&15; VGPR j holds K = 2j + (j>=4?8:0) + 8g, and +1
// ---------------------------------------------------------------------------
template<bool TAIL>
__device__ __forceinline__ void proj_step(
    int v0, int vend, int f0, int lane,
    const float* __restrict__ fp, const float* __restrict__ ep,
    v8f* acc0, v8f* acc1) {
  const int m = lane & 15;
  const int g = lane >> 4;

  v16bf ahi0, alo0, ahi1, alo1;
#pragma unroll
  for (int j = 0; j < 8; ++j) {
    const int k0 = 2 * j + (j >= 4 ? 8 : 0) + 8 * g;
    float a00 = 0.f, a01 = 0.f, a10 = 0.f, a11 = 0.f;
    if (!TAIL || (v0 + k0 < vend)) {
      const size_t r0 = (size_t)(v0 + k0) * NF_ + f0 + m;
      a00 = fp[r0];
      a10 = fp[r0 + 16];
    }
    if (!TAIL || (v0 + k0 + 1 < vend)) {
      const size_t r1 = (size_t)(v0 + k0 + 1) * NF_ + f0 + m;
      a01 = fp[r1];
      a11 = fp[r1 + 16];
    }
    __bf16 h, l;
    bf16_split(a00, h, l); ahi0[2*j]   = h; alo0[2*j]   = l;
    bf16_split(a01, h, l); ahi0[2*j+1] = h; alo0[2*j+1] = l;
    bf16_split(a10, h, l); ahi1[2*j]   = h; alo1[2*j]   = l;
    bf16_split(a11, h, l); ahi1[2*j+1] = h; alo1[2*j+1] = l;
  }

#pragma unroll
  for (int nt = 0; nt < 8; ++nt) {
    const int n    = nt * 16 + m;
    const bool nok = (nt < 7) || (m < K_ - 7 * 16);   // n < 120
    v16bf bhi, blo;
#pragma unroll
    for (int j = 0; j < 8; ++j) {
      const int k0 = 2 * j + (j >= 4 ? 8 : 0) + 8 * g;
      float b0 = 0.f, b1 = 0.f;
      if (nok && (!TAIL || (v0 + k0     < vend))) b0 = ep[(size_t)(v0 + k0    ) * K_ + n];
      if (nok && (!TAIL || (v0 + k0 + 1 < vend))) b1 = ep[(size_t)(v0 + k0 + 1) * K_ + n];
      __bf16 h, l;
      bf16_split(b0, h, l); bhi[2*j]   = h; blo[2*j]   = l;
      bf16_split(b1, h, l); bhi[2*j+1] = h; blo[2*j+1] = l;
    }
    acc0[nt] = __builtin_amdgcn_wmma_f32_16x16x32_bf16(false, ahi0, false, bhi, (short)0, acc0[nt], false, false);
    acc0[nt] = __builtin_amdgcn_wmma_f32_16x16x32_bf16(false, ahi0, false, blo, (short)0, acc0[nt], false, false);
    acc0[nt] = __builtin_amdgcn_wmma_f32_16x16x32_bf16(false, alo0, false, bhi, (short)0, acc0[nt], false, false);
    acc1[nt] = __builtin_amdgcn_wmma_f32_16x16x32_bf16(false, ahi1, false, bhi, (short)0, acc1[nt], false, false);
    acc1[nt] = __builtin_amdgcn_wmma_f32_16x16x32_bf16(false, ahi1, false, blo, (short)0, acc1[nt], false, false);
    acc1[nt] = __builtin_amdgcn_wmma_f32_16x16x32_bf16(false, alo1, false, bhi, (short)0, acc1[nt], false, false);
  }
}

__global__ __launch_bounds__(32) void proj_kernel(
    const float* __restrict__ feat, const float* __restrict__ evecs,
    float* __restrict__ partial) {
  const int mp    = blockIdx.x;          // 0..10  (pair of M tiles)
  const int b     = blockIdx.y;          // 0..15  (batch)
  const int chunk = blockIdx.z;          // 0..NSPLIT-1 (V chunk)
  const int lane  = threadIdx.x;
  const float* fp = feat  + (size_t)b * V_ * NF_;
  const float* ep = evecs + (size_t)b * V_ * K_;
  const int f0 = mp * 32;

  constexpr int CH = V_ / NSPLIT;        // 1250
  const int vs = chunk * CH;
  const int ve = vs + CH;

  v8f zero = {};
  v8f acc0[8], acc1[8];
#pragma unroll
  for (int t = 0; t < 8; ++t) { acc0[t] = zero; acc1[t] = zero; }

  int v0 = vs;
  for (; v0 + 32 <= ve; v0 += 32) {
    // prefetch next step's A tile: lane -> row, 128B row segment = whole pair width
    if (v0 + 32 + lane < V_)
      __builtin_prefetch(fp + (size_t)(v0 + 32 + lane) * NF_ + f0, 0, 1);
    proj_step<false>(v0, ve, f0, lane, fp, ep, acc0, acc1);
  }
  if (v0 < ve)
    proj_step<true>(v0, ve, f0, lane, fp, ep, acc0, acc1);

  // C/D layout: VGPR r -> (M=r, N=lane) lanes0-15 ; (M=r+8, N=lane-16) lanes16-31
  float* pc = partial + ((size_t)chunk * B_ + b) * NF_ * KP_;
  const int m = lane & 15, g = lane >> 4;
#pragma unroll
  for (int nt = 0; nt < 8; ++nt) {
#pragma unroll
    for (int r = 0; r < 8; ++r) {
      const int mm = f0 + r + 8 * g;
      const int nn = nt * 16 + m;
      pc[(size_t)mm * KP_ + nn]        = acc0[nt][r];   // padding cols = exact 0
      pc[(size_t)(mm + 16) * KP_ + nn] = acc1[nt][r];
    }
  }
}

// ---------------------------------------------------------------------------
// Kernel 1b: deterministic split-K reduction (no float atomics).
// ---------------------------------------------------------------------------
__global__ __launch_bounds__(256) void reduce_kernel(
    const float* __restrict__ partial, float* __restrict__ out) {
  constexpr size_t STRIDE4 = (size_t)B_ * NF_ * KP_ / 4;
  const size_t i = (size_t)blockIdx.x * blockDim.x + threadIdx.x;
  if (i < STRIDE4) {
    const float4* p4 = (const float4*)partial;
    float4 s = p4[i];
#pragma unroll
    for (int c = 1; c < NSPLIT; ++c) {
      const float4 v = p4[(size_t)c * STRIDE4 + i];
      s.x += v.x; s.y += v.y; s.z += v.z; s.w += v.w;
    }
    ((float4*)out)[i] = s;
  }
}

// ---------------------------------------------------------------------------
// Kernel 2: Gram matrices in full fp32 via v_wmma_f32_16x16x4_f32.
// which: 0 -> FtF, 1 -> FtG, 2 -> GtG.  (GtF = FtG^T, derived in the solver.)
// f32 A 16x4 layout: VGPR j holds K = j + 2g ; B 4x16 mirrors it.
// ---------------------------------------------------------------------------
__global__ __launch_bounds__(32) void gram_kernel(
    const float* __restrict__ fhat, const float* __restrict__ ghat,
    float* __restrict__ grams) {
  const int mt    = blockIdx.x & 7;
  const int nt    = blockIdx.x >> 3;
  const int b     = blockIdx.y;
  const int which = blockIdx.z;
  const float* Ab = ((which == 2) ? ghat : fhat) + (size_t)b * NF_ * KP_;
  const float* Bb = ((which == 0) ? fhat : ghat) + (size_t)b * NF_ * KP_;
  const int lane = threadIdx.x;
  const int m = lane & 15, g = lane >> 4;

  v8f acc = {};
  for (int f = 0; f < NF_; f += 4) {
    v2f av, bv;
#pragma unroll
    for (int j = 0; j < 2; ++j) {
      const int fr = f + j + 2 * g;
      av[j] = Ab[(size_t)fr * KP_ + mt * 16 + m];
      bv[j] = Bb[(size_t)fr * KP_ + nt * 16 + m];
    }
    acc = __builtin_amdgcn_wmma_f32_16x16x4_f32(false, av, false, bv, (short)0, acc, false, false);
  }

  float* gp = grams + ((size_t)which * B_ + b) * K_ * K_;
#pragma unroll
  for (int r = 0; r < 8; ++r) {
    const int mm = mt * 16 + r + 8 * g;
    const int nn = nt * 16 + m;
    if (mm < K_ && nn < K_) gp[mm * K_ + nn] = acc[r];
  }
}

// ---------------------------------------------------------------------------
// Kernel 3: batched 120x120 solve + transpose, all in LDS.
// blockIdx.x = b*2 + which (0: C1 = solve(FtF,FtG)^T, 1: C2 = solve(GtG,FtG^T)^T)
// SPD Gram matrices -> Gaussian elimination without pivoting is stable.
// Forward elimination: even/odd thread pair per row splits A-update / RHS-update.
// ---------------------------------------------------------------------------
__global__ __launch_bounds__(256) void solve_kernel(
    const float* __restrict__ grams, float* __restrict__ out) {
  extern __shared__ float sm[];
  constexpr int S = K_ + 1;              // padded row stride -> no bank conflicts
  float* A  = sm;                        // 120 x 121
  float* Bm = sm + K_ * S;               // 120 x 121
  const int b     = blockIdx.x >> 1;
  const int which = blockIdx.x & 1;
  const int t = threadIdx.x;

  const float* FtF = grams + ((size_t)0 * B_ + b) * K_ * K_;
  const float* FtG = grams + ((size_t)1 * B_ + b) * K_ * K_;
  const float* GtG = grams + ((size_t)2 * B_ + b) * K_ * K_;

  for (int i = t; i < K_ * K_; i += 256) {
    const int r = i / K_, c = i % K_;
    A[r * S + c]  = (which == 0) ? FtF[i] : GtG[i];
    Bm[r * S + c] = (which == 0) ? FtG[i] : FtG[c * K_ + r];   // GtF = FtG^T
  }
  __syncthreads();

  // forward elimination
  for (int col = 0; col < K_ - 1; ++col) {
    const float pinv = 1.0f / A[col * S + col];
    const int r = col + 1 + (t >> 1);
    if (r < K_) {
      const float fac = A[r * S + col] * pinv;     // A[r][col] untouched this step
      if ((t & 1) == 0) {
        for (int j = col + 1; j < K_; ++j) A[r * S + j]  -= fac * A[col * S + j];
      } else {
        for (int j = 0; j < K_; ++j)       Bm[r * S + j] -= fac * Bm[col * S + j];
      }
    }
    __syncthreads();
  }

  // back substitution: thread t owns column t of X (in-place in Bm) -> no barriers
  if (t < K_) {
    for (int r = K_ - 1; r >= 0; --r) {
      float x = Bm[r * S + t];
      for (int p = r + 1; p < K_; ++p) x -= A[r * S + p] * Bm[p * S + t];
      Bm[r * S + t] = x / A[r * S + r];
    }
  }
  __syncthreads();

  // C = X^T
  float* o = out + ((size_t)which * B_ + b) * K_ * K_;
  for (int i = t; i < K_ * K_; i += 256) {
    const int r = i / K_, c = i % K_;
    o[r * K_ + c] = Bm[c * S + r];
  }
}

// ---------------------------------------------------------------------------
extern "C" void kernel_launch(void* const* d_in, const int* in_sizes, int n_in,
                              void* d_out, int out_size, void* d_ws, size_t ws_size,
                              hipStream_t stream) {
  (void)in_sizes; (void)n_in; (void)out_size; (void)ws_size;
  const float* feat_x  = (const float*)d_in[0];
  const float* feat_y  = (const float*)d_in[1];
  const float* evecs_x = (const float*)d_in[2];
  const float* evecs_y = (const float*)d_in[3];
  float* out = (float*)d_out;

  constexpr size_t FH = (size_t)B_ * NF_ * KP_;      // 720,896 floats
  float* pf    = (float*)d_ws;                       // [NSPLIT][B][NF][KP]
  float* pg    = pf + NSPLIT * FH;                   // [NSPLIT][B][NF][KP]
  float* fhat  = pg + NSPLIT * FH;                   // [B][NF][KP]
  float* ghat  = fhat + FH;                          // [B][NF][KP]
  float* grams = ghat + FH;                          // [3][B][K][K]

  dim3 gproj(NF_ / 32, B_, NSPLIT);
  proj_kernel<<<gproj, 32, 0, stream>>>(feat_x, evecs_x, pf);
  proj_kernel<<<gproj, 32, 0, stream>>>(feat_y, evecs_y, pg);

  const int nred = (int)((FH / 4 + 255) / 256);
  reduce_kernel<<<nred, 256, 0, stream>>>(pf, fhat);
  reduce_kernel<<<nred, 256, 0, stream>>>(pg, ghat);

  dim3 ggram(64, B_, 3);
  gram_kernel<<<ggram, 32, 0, stream>>>(fhat, ghat, grams);

  const size_t lds = (size_t)2 * K_ * (K_ + 1) * sizeof(float);
  solve_kernel<<<B_ * 2, 256, lds, stream>>>(grams, out);
}
